// ElementTransformer_24197845746070
// MI455X (gfx1250) — compile-verified, matrix operand order checked by reading, not run on hardware
//
#include <hip/hip_runtime.h>
#include <hip/hip_bf16.h>

// ---------------------------------------------------------------------------
// ElementTransformer on MI455X (gfx1250). All dense contractions via
// v_wmma_f32_16x16x32_f16. Weights pre-transposed/converted to f16 so every
// fragment load is an unconditional 16B vector load (no guards, no per-element
// waits). Fragment layout per CDNA5 16-bit A/B spec: lane half hi reads two
// contiguous 8-half runs at k0+hi*8 and k0+16+hi*8.
// ---------------------------------------------------------------------------

typedef __attribute__((ext_vector_type(16))) _Float16 v16h;
typedef __attribute__((ext_vector_type(8)))  _Float16 v8h;
typedef __attribute__((ext_vector_type(8)))  float    v8f;

__device__ __forceinline__ v8f wmma_16x16x32(v16h a, v16h b, v8f c) {
  return __builtin_amdgcn_wmma_f32_16x16x32_f16(false, a, false, b, (short)0, c,
                                                false, false);
}

__device__ __forceinline__ v16h cat16(v8h a, v8h b) {
  return __builtin_shufflevector(a, b, 0, 1, 2, 3, 4, 5, 6, 7, 8, 9, 10, 11, 12,
                                 13, 14, 15);
}

__device__ __forceinline__ float silu_f(float x) {
  return x * (1.0f / (1.0f + __expf(-x)));
}

// ---------------------------------------------------------------------------
// Edge featurization (CL=0, CU=5, ALPHA=1). ea is [M,64], cols 50..63 zero.
// ---------------------------------------------------------------------------
__global__ __launch_bounds__(256)
void edge_kernel(const float* __restrict__ pos, const int* __restrict__ ei,
                 const int* __restrict__ z, const float* __restrict__ means,
                 const float* __restrict__ betas, float* __restrict__ ea,
                 float* __restrict__ C, float* __restrict__ Cn,
                 int* __restrict__ seg, int M, int E, int R) {
  int m = blockIdx.x * 256 + threadIdx.x;
  if (m >= M) return;
  int s = ei[m], d = ei[M + m];
  float dx = pos[d * 3 + 0] - pos[s * 3 + 0];
  float dy = pos[d * 3 + 1] - pos[s * 3 + 1];
  float dz = pos[d * 3 + 2] - pos[s * 3 + 2];
  float dist = sqrtf(dx * dx + dy * dy + dz * dz);
  const float PI = 3.14159265358979323846f;
  float c = (dist < 5.0f) ? 0.5f * (__cosf(dist * (PI / 5.0f)) + 1.0f) : 0.0f;
  C[m] = c;
  Cn[m] = (s != d) ? c : 0.0f;
  seg[m] = d * E + z[s];
  float ex = __expf(-dist);
  for (int r = 0; r < 64; ++r) {
    float v = 0.0f;
    if (r < R) {
      float t = ex - means[r];
      v = c * __expf(-betas[r] * t * t);
    }
    ea[(size_t)m * 64 + r] = v;
  }
}

__global__ __launch_bounds__(256)
void fill_zero(float* __restrict__ p, long n) {
  long i = (long)blockIdx.x * 256 + threadIdx.x;
  if (i < n) p[i] = 0.0f;
}

__global__ __launch_bounds__(256)
void gather_emb(const float* __restrict__ emb, const int* __restrict__ z,
                float* __restrict__ xe, int N) {
  int i = blockIdx.x * 256 + threadIdx.x;
  if (i >= N * 128) return;
  int n = i >> 7, f = i & 127;
  xe[i] = emb[(size_t)z[n] * 128 + f];
}

__global__ __launch_bounds__(256)
void set_mask(const int* __restrict__ seg, float* __restrict__ maskf, int M) {
  int m = blockIdx.x * 256 + threadIdx.x;
  if (m < M) maskf[seg[m]] = 1.0f;
}

// Transpose + convert weight [K,128] f32 -> Wt[128][Kc] f16, zero pad k>=K.
__global__ __launch_bounds__(256)
void prep_wt(const float* __restrict__ W, _Float16* __restrict__ Wt, int K,
             int Kc) {
  int idx = blockIdx.x * 256 + threadIdx.x;
  if (idx >= 128 * Kc) return;
  int col = idx / Kc, k = idx - col * Kc;
  Wt[idx] = (k < K) ? (_Float16)W[(size_t)k * 128 + col] : (_Float16)0.0f;
}

__global__ __launch_bounds__(256)
void scatter_nbr(const float* __restrict__ nbr_emb, const int* __restrict__ z,
                 const int* __restrict__ ei, const float* __restrict__ Wn,
                 float* __restrict__ xn, int M) {
  int t = blockIdx.x * 256 + threadIdx.x;
  int m = t >> 7, f = t & 127;
  if (m >= M) return;
  int s = ei[m], d = ei[M + m];
  atomicAdd(&xn[(size_t)d * 128 + f],
            nbr_emb[(size_t)z[s] * 128 + f] * Wn[(size_t)m * 128 + f]);
}

__global__ __launch_bounds__(256)
void scatter_msg(const float* __restrict__ h, const float* __restrict__ W,
                 const int* __restrict__ ei, const int* __restrict__ seg,
                 float* __restrict__ y, int M) {
  int t = blockIdx.x * 256 + threadIdx.x;
  int m = t >> 7, f = t & 127;
  if (m >= M) return;
  int s = ei[m];
  atomicAdd(&y[(size_t)seg[m] * 128 + f],
            h[(size_t)s * 128 + f] * W[(size_t)m * 128 + f]);
}

// ---------------------------------------------------------------------------
// WMMA GEMM: out[rows,128] = epi(A[rows,lda](:, :KC) @ Wt^T), Wt f16 [128][KC].
// rows must be a multiple of 16 (true here: M=320000, N=20000).
// Block: 256 thr = 8 waves; wave w owns columns [16w,16w+16); block owns a
// 16-row stripe staged once into LDS as f16.
// ---------------------------------------------------------------------------
template <int KC>
__global__ __launch_bounds__(256)
void gemm128_wmma(const float* __restrict__ A, int lda,
                  const _Float16* __restrict__ Wt,
                  const float* __restrict__ bias,
                  const float* __restrict__ rowscale, float* __restrict__ out,
                  int act, int addto) {
  __shared__ __align__(16) _Float16 As[16][KC + 8];  // 16B-aligned rows
  const int tid = threadIdx.x;
  const int r0 = blockIdx.x * 16;
  // stage A tile (unguarded, float4-vectorized)
  for (int idx = tid; idx < 4 * KC; idx += 256) {
    int r = idx / (KC / 4), c4 = idx - r * (KC / 4);
    const float4 v = *(const float4*)(A + (size_t)(r0 + r) * lda + c4 * 4);
    As[r][c4 * 4 + 0] = (_Float16)v.x;
    As[r][c4 * 4 + 1] = (_Float16)v.y;
    As[r][c4 * 4 + 2] = (_Float16)v.z;
    As[r][c4 * 4 + 3] = (_Float16)v.w;
  }
  __syncthreads();

  const int wv = tid >> 5, lane = tid & 31;
  const int lh = lane & 15, hi = lane >> 4;
  const int col = wv * 16 + lh;
  const _Float16* __restrict__ Wc = Wt + (size_t)col * KC;

  v8f acc = {};
#pragma unroll
  for (int k0 = 0; k0 < KC; k0 += 32) {
    v8h a0 = *(const v8h*)&As[lh][k0 + hi * 8];
    v8h a1 = *(const v8h*)&As[lh][k0 + 16 + hi * 8];
    v8h b0 = *(const v8h*)(Wc + k0 + hi * 8);
    v8h b1 = *(const v8h*)(Wc + k0 + 16 + hi * 8);
    acc = wmma_16x16x32(cat16(a0, a1), cat16(b0, b1), acc);
  }

  float bcol = bias ? bias[col] : 0.0f;
#pragma unroll
  for (int v = 0; v < 8; ++v) {
    int row = r0 + v + (hi << 3);
    float t = acc[v] + bcol;
    if (act) t = silu_f(t);
    if (rowscale) t *= rowscale[row];
    size_t o = (size_t)row * 128 + col;
    if (addto) t += out[o];
    out[o] = t;
  }
}

// ---------------------------------------------------------------------------
// Fused per-node element attention (one block = one node, 8 waves = 8 heads /
// 8 column tiles). E=10 padded to 16; DH=16 padded to K=32 with zeros.
// ---------------------------------------------------------------------------
__global__ __launch_bounds__(256)
void attn_fused(const float* __restrict__ y, const float* __restrict__ maskf,
                const _Float16* __restrict__ qwT, const float* __restrict__ qb,
                const _Float16* __restrict__ kwT, const float* __restrict__ kb,
                const _Float16* __restrict__ vwT, const float* __restrict__ vb,
                const _Float16* __restrict__ o1wT, const float* __restrict__ o1b,
                float* __restrict__ node, int E) {
  __shared__ __align__(16) _Float16 yl[16][136];
  __shared__ __align__(16) _Float16 qs[16][136];
  __shared__ __align__(16) _Float16 ks[16][136];
  __shared__ __align__(16) _Float16 vT[128][24];  // v transposed: vT[d][f]
  __shared__ __align__(16) _Float16 ats[16][136];
  __shared__ __align__(16) _Float16 os[16][136];
  __shared__ float mk[16];

  const int n = blockIdx.x;
  const int tid = threadIdx.x;
  if (tid < 16) mk[tid] = (tid < E) ? maskf[(size_t)n * E + tid] : 0.0f;
  for (int idx = tid; idx < 16 * 32; idx += 256) {  // float4 granularity
    int e = idx >> 5, c4 = idx & 31;
    float4 v = {0.0f, 0.0f, 0.0f, 0.0f};
    if (e < E) v = *(const float4*)(y + ((size_t)n * E + e) * 128 + c4 * 4);
    yl[e][c4 * 4 + 0] = (_Float16)v.x;
    yl[e][c4 * 4 + 1] = (_Float16)v.y;
    yl[e][c4 * 4 + 2] = (_Float16)v.z;
    yl[e][c4 * 4 + 3] = (_Float16)v.w;
  }
  __syncthreads();

  const int wv = tid >> 5, lane = tid & 31;
  const int lh = lane & 15, hi = lane >> 4;
  const int col = wv * 16 + lh;
  const int h = wv;
  const v8h zero8 = {};

  // ---- q, k, v projections (K=128) ----------------------------------------
#pragma unroll
  for (int m = 0; m < 3; ++m) {
    const _Float16* Wm = (m == 0) ? qwT : (m == 1) ? kwT : vwT;
    const float* Bm = (m == 0) ? qb : (m == 1) ? kb : vb;
    const _Float16* __restrict__ Wc = Wm + (size_t)col * 128;
    v8f acc = {};
#pragma unroll
    for (int k0 = 0; k0 < 128; k0 += 32) {
      v8h a0 = *(const v8h*)&yl[lh][k0 + hi * 8];
      v8h a1 = *(const v8h*)&yl[lh][k0 + 16 + hi * 8];
      v8h b0 = *(const v8h*)(Wc + k0 + hi * 8);
      v8h b1 = *(const v8h*)(Wc + k0 + 16 + hi * 8);
      acc = wmma_16x16x32(cat16(a0, a1), cat16(b0, b1), acc);
    }
    float bcol = Bm[col];
    if (m == 0) {
#pragma unroll
      for (int v = 0; v < 8; ++v) qs[v + (hi << 3)][col] = (_Float16)(acc[v] + bcol);
    } else if (m == 1) {
#pragma unroll
      for (int v = 0; v < 8; ++v) ks[v + (hi << 3)][col] = (_Float16)(acc[v] + bcol);
    } else {
#pragma unroll
      for (int v = 0; v < 8; ++v) vT[col][v + (hi << 3)] = (_Float16)(acc[v] + bcol);
    }
  }
  __syncthreads();

  // ---- scores_h = q_h @ k_h^T ; attn = silu * mask_e * mask_f -------------
  {
    v8h qa = *(const v8h*)&qs[lh][h * 16 + hi * 8];
    v8h kb8 = *(const v8h*)&ks[lh][h * 16 + hi * 8];  // B[k][f] = ks[f][h*16+k]
    v8f acc = {};
    acc = wmma_16x16x32(cat16(qa, zero8), cat16(kb8, zero8), acc);
    float mcol = mk[lh];
#pragma unroll
    for (int v = 0; v < 8; ++v) {
      int e = v + (hi << 3);
      ats[e][h * 16 + lh] = (_Float16)(silu_f(acc[v]) * mk[e] * mcol);
    }
  }
  __syncthreads();

  // ---- o_h = attn_h @ v_h --------------------------------------------------
  {
    v8h aa = *(const v8h*)&ats[lh][h * 16 + hi * 8];
    v8h bb = *(const v8h*)&vT[h * 16 + lh][hi * 8];  // B[k][d] = vT[h*16+d][k]
    v8f acc = {};
    acc = wmma_16x16x32(cat16(aa, zero8), cat16(bb, zero8), acc);
#pragma unroll
    for (int v = 0; v < 8; ++v) os[v + (hi << 3)][h * 16 + lh] = (_Float16)acc[v];
  }
  __syncthreads();

  // ---- out2 = o @ o1w + o1b ; node = sum_e mk[e]*out2[e,col] ---------------
  {
    const _Float16* __restrict__ Wc = o1wT + (size_t)col * 128;
    v8f acc = {};
#pragma unroll
    for (int k0 = 0; k0 < 128; k0 += 32) {
      v8h a0 = *(const v8h*)&os[lh][k0 + hi * 8];
      v8h a1 = *(const v8h*)&os[lh][k0 + 16 + hi * 8];
      v8h b0 = *(const v8h*)(Wc + k0 + hi * 8);
      v8h b1 = *(const v8h*)(Wc + k0 + 16 + hi * 8);
      acc = wmma_16x16x32(cat16(a0, a1), cat16(b0, b1), acc);
    }
    float bcol = o1b[col];
    float partial = 0.0f;
#pragma unroll
    for (int v = 0; v < 8; ++v) partial += (acc[v] + bcol) * mk[v + (hi << 3)];
    partial += __shfl_xor(partial, 16, 32);
    if (hi == 0) node[(size_t)n * 128 + col] = partial;
  }
}

// ---------------------------------------------------------------------------
// Host launch sequence
// ---------------------------------------------------------------------------
extern "C" void kernel_launch(void* const* d_in, const int* in_sizes, int n_in,
                              void* d_out, int out_size, void* d_ws,
                              size_t ws_size, hipStream_t stream) {
  const int N = in_sizes[0];
  const int M = in_sizes[2] / 2;
  const int E = 10, F = 128, R = 50, L = 6;

  const int*   z       = (const int*)d_in[0];
  const float* pos     = (const float*)d_in[1];
  const int*   ei      = (const int*)d_in[2];
  const float* emb     = (const float*)d_in[3];
  const float* nbr_emb = (const float*)d_in[4];
  const float* nbr_dw  = (const float*)d_in[5];
  const float* nbr_db  = (const float*)d_in[6];
  const float* nbr_cw  = (const float*)d_in[7];
  const float* nbr_cb  = (const float*)d_in[8];
  const float* means   = (const float*)d_in[9];
  const float* betas   = (const float*)d_in[10];
  const float* lin1_w  = (const float*)d_in[11];
  const float* fn1_w   = (const float*)d_in[12];
  const float* fn1_b   = (const float*)d_in[13];
  const float* fn2_w   = (const float*)d_in[14];
  const float* fn2_b   = (const float*)d_in[15];
  const float* qw      = (const float*)d_in[16];
  const float* qb      = (const float*)d_in[17];
  const float* kw      = (const float*)d_in[18];
  const float* kb      = (const float*)d_in[19];
  const float* vw      = (const float*)d_in[20];
  const float* vb      = (const float*)d_in[21];
  const float* o1w     = (const float*)d_in[22];
  const float* o1b     = (const float*)d_in[23];
  const float* o2w     = (const float*)d_in[24];
  const float* o2b     = (const float*)d_in[25];

  float* x = (float*)d_out;  // [N,128]

  // f32 workspace; y aliases tmp1 (disjoint lifetimes), h=xe, node=xn.
  float* ws = (float*)d_ws;
  size_t off = 0;
  float* ea    = ws + off; off += (size_t)M * 64;
  float* Cw    = ws + off; off += M;
  float* Cn    = ws + off; off += M;
  int*   seg   = (int*)(ws + off); off += M;
  float* maskf = ws + off; off += (size_t)N * E;
  float* xe    = ws + off; off += (size_t)N * F;
  float* xn    = ws + off; off += (size_t)N * F;
  float* tmp1  = ws + off; off += (size_t)M * F;
  float* Wb    = ws + off; off += (size_t)M * F;
  float* h    = xe;
  float* node = xn;
  float* y    = tmp1;

  // f16 transposed-weight pool
  const int S64 = 128 * 64, S128 = 128 * 128;
  _Float16* hp = (_Float16*)(ws + off);
  size_t ho = 0;
  _Float16* dwT  = hp + ho; ho += S64;
  _Float16* cw1T = hp + ho; ho += S128;
  _Float16* cw2T = hp + ho; ho += S128;
  _Float16* layT = hp + ho;  // per layer: fn1T,fn2T,lin1T,qT,kT,vT,o1T,o2T
  const size_t LSTRIDE = (size_t)S64 + 7 * S128;

  dim3 blk(256);
  const int gM   = (M + 255) / 256;
  const int gNF  = (N * F + 255) / 256;
  const int gMF  = (int)(((size_t)M * F + 255) / 256);
  const int gNE  = (N * E + 255) / 256;
  const int gNEF = (int)(((size_t)N * E * F + 255) / 256);
  const int g64  = (S64 + 255) / 256, g128 = (S128 + 255) / 256;
  const int gemm_gM = M / 16, gemm_gN = N / 16;

  // ---- weight prep (f16, transposed, K padded to 32) ----------------------
  prep_wt<<<g64, blk, 0, stream>>>(nbr_dw, dwT, R, 64);
  prep_wt<<<g128, blk, 0, stream>>>(nbr_cw, cw1T, F, 128);
  prep_wt<<<g128, blk, 0, stream>>>(nbr_cw + (size_t)F * F, cw2T, F, 128);
  for (int l = 0; l < L; ++l) {
    _Float16* lb = layT + l * LSTRIDE;
    prep_wt<<<g64, blk, 0, stream>>>(fn1_w + (size_t)l * R * F, lb, R, 64);
    prep_wt<<<g128, blk, 0, stream>>>(fn2_w + (size_t)l * F * F, lb + S64, F, 128);
    prep_wt<<<g128, blk, 0, stream>>>(lin1_w + (size_t)l * F * F, lb + S64 + S128, F, 128);
    prep_wt<<<g128, blk, 0, stream>>>(qw + (size_t)l * F * F, lb + S64 + 2 * S128, F, 128);
    prep_wt<<<g128, blk, 0, stream>>>(kw + (size_t)l * F * F, lb + S64 + 3 * S128, F, 128);
    prep_wt<<<g128, blk, 0, stream>>>(vw + (size_t)l * F * F, lb + S64 + 4 * S128, F, 128);
    prep_wt<<<g128, blk, 0, stream>>>(o1w + (size_t)l * F * F, lb + S64 + 5 * S128, F, 128);
    prep_wt<<<g128, blk, 0, stream>>>(o2w + (size_t)l * F * F, lb + S64 + 6 * S128, F, 128);
  }

  // ---- edge features, masks, neighbor embedding ---------------------------
  edge_kernel<<<gM, blk, 0, stream>>>(pos, ei, z, means, betas, ea, Cw, Cn, seg, M, E, R);
  fill_zero<<<gNF, blk, 0, stream>>>(xn, (long)N * F);
  fill_zero<<<gNE, blk, 0, stream>>>(maskf, (long)N * E);
  gather_emb<<<gNF, blk, 0, stream>>>(emb, z, xe, N);
  set_mask<<<gM, blk, 0, stream>>>(seg, maskf, M);

  // Wn = (ea @ nbr_dw + nbr_db) * C * notself
  gemm128_wmma<64><<<gemm_gM, blk, 0, stream>>>(ea, 64, dwT, nbr_db, Cn, Wb, 0, 0);
  scatter_nbr<<<gMF, blk, 0, stream>>>(nbr_emb, z, ei, Wb, xn, M);
  // x = [xe, xn] @ nbr_cw + nbr_cb
  gemm128_wmma<128><<<gemm_gN, blk, 0, stream>>>(xe, 128, cw1T, nullptr, nullptr, x, 0, 0);
  gemm128_wmma<128><<<gemm_gN, blk, 0, stream>>>(xn, 128, cw2T, nbr_cb, nullptr, x, 0, 1);

  // ---- layers -------------------------------------------------------------
  for (int l = 0; l < L; ++l) {
    _Float16* lb = layT + l * LSTRIDE;
    const size_t oF = (size_t)l * F;
    // W = (silu(ea@fn1+b1) @ fn2 + b2) * C
    gemm128_wmma<64><<<gemm_gM, blk, 0, stream>>>(ea, 64, lb, fn1_b + oF, nullptr, tmp1, 1, 0);
    gemm128_wmma<128><<<gemm_gM, blk, 0, stream>>>(tmp1, 128, lb + S64, fn2_b + oF, Cw, Wb, 0, 0);
    // h = x @ lin1
    gemm128_wmma<128><<<gemm_gN, blk, 0, stream>>>(x, 128, lb + S64 + S128, nullptr, nullptr, h, 0, 0);
    // y = segment_sum(h[src]*W, seg)
    fill_zero<<<gNEF, blk, 0, stream>>>(y, (long)N * E * F);
    scatter_msg<<<gMF, blk, 0, stream>>>(h, Wb, ei, seg, y, M);
    // fused element attention -> node
    attn_fused<<<N, blk, 0, stream>>>(y, maskf, lb + S64 + 2 * S128, qb + oF,
                                      lb + S64 + 3 * S128, kb + oF,
                                      lb + S64 + 4 * S128, vb + oF,
                                      lb + S64 + 5 * S128, o1b + oF, node, E);
    // x += silu(node @ o2w + o2b)
    gemm128_wmma<128><<<gemm_gN, blk, 0, stream>>>(node, 128, lb + S64 + 6 * S128,
                                                   o2b + oF, nullptr, x, 1, 1);
  }
}
